// VisionModel_42511586296291
// MI455X (gfx1250) — matrix operand order repaired and checked by reading, not compiled, not run on hardware
//
#include <hip/hip_runtime.h>
#include <hip/hip_bf16.h>
#include <math.h>

typedef __bf16 bf16;
typedef __attribute__((ext_vector_type(16))) __bf16 v16bf;
typedef __attribute__((ext_vector_type(8)))  float  v8f;

#define S_TOK 1024
#define DMODEL 1152
#define NHEAD 16
#define HDIM 72
#define HDIM_P 96
#define FFDIM 4304
#define MERGED 4608
#define OUTD 3584

// ---------------- GEMM: C = A(bf16, MxK) * B(bf16, KxN) [+bias][gelu][+res] ----------------
// 256 threads = 8 wave32, block tile 128x64x32, each wave computes 32x32 (2x2 WMMA frags).
#define BM 128
#define BN 64
#define BK 32
#define LDA_S 40   // BK + 8 bf16 pad -> 80B rows (16B aligned)
#define LDB_S 40   // Bs is stored transposed: [BN][BK+8]

__device__ __forceinline__ float gelu_tanh(float x) {
    float x3 = x * x * x;
    return 0.5f * x * (1.0f + tanhf(0.7978845608f * (x + 0.044715f * x3)));
}

__launch_bounds__(256, 2)
__global__ void gemm_bf16_wmma(const bf16* __restrict__ A, int lda,
                               const bf16* __restrict__ B, int ldb, int transB,
                               const float* __restrict__ bias,
                               const float* res, int ldres,
                               float* outF, bf16* outH, int ldc,
                               int M, int N, int K, int do_gelu)
{
    __shared__ bf16 As[BM * LDA_S];   // row-major: [m][k]
    __shared__ bf16 Bs[BN * LDB_S];   // transposed: [n][k]

    const int tid  = threadIdx.x;
    const int lane = tid & 31;
    const int wave = tid >> 5;        // 0..7
    const int half = lane >> 4;       // lane group 0/1
    const int l16  = lane & 15;
    const int rowBase = blockIdx.y * BM;
    const int colBase = blockIdx.x * BN;
    const int wm = (wave >> 1) * 32;  // 0,32,64,96
    const int wn = (wave & 1) * 32;   // 0,32

    // staging indices (K-loop invariant)
    const int a_r    = tid >> 1;                 // A tile row 0..127
    const int a_cseg = (tid & 1) * 16;           // A k segment
    const int a_gr   = rowBase + a_r;
    const int a_grc  = (a_gr < M) ? a_gr : (M - 1);     // clamped (always legal)
    const int bn_k   = tid >> 3;                 // non-trans: k row 0..31
    const int bn_ns  = (tid & 7) * 8;            // non-trans: n segment
    const int bt_n   = tid >> 2;                 // trans: n 0..63
    const int bt_ks  = (tid & 3) * 8;            // trans: k segment

    v8f acc[2][2];
    #pragma unroll
    for (int i = 0; i < 2; ++i)
        #pragma unroll
        for (int j = 0; j < 2; ++j)
            #pragma unroll
            for (int r = 0; r < 8; ++r) acc[i][j][r] = 0.0f;

    for (int k0 = 0; k0 < K; k0 += BK) {
        // ---- stage A tile: 128 rows x 32 cols ----
        {
            int gc = k0 + a_cseg;
            bf16* dst = &As[a_r * LDA_S + a_cseg];
            if (a_gr < M && gc + 16 <= K) {
                const uint4* src = reinterpret_cast<const uint4*>(&A[(size_t)a_gr * lda + gc]);
                *reinterpret_cast<uint4*>(dst)     = src[0];
                *reinterpret_cast<uint4*>(dst + 8) = src[1];
                if (k0 + BK < K)
                    __builtin_prefetch(&A[(size_t)a_gr * lda + gc + BK], 0, 1);
            } else {
                // branch-free edge: clamped address load + select-zero
                #pragma unroll
                for (int e = 0; e < 16; ++e) {
                    int gce = gc + e;
                    int gcc = (gce < K) ? gce : (K - 1);
                    bf16 v = A[(size_t)a_grc * lda + gcc];
                    dst[e] = (a_gr < M && gce < K) ? v : (bf16)0.0f;
                }
            }
        }
        // ---- stage B tile into transposed LDS layout Bs[n][k] ----
        if (!transB) {
            // global B is [K][N] row-major: coalesced b128 reads along n, scattered u16 LDS writes
            int gk = k0 + bn_k;
            int gn = colBase + bn_ns;
            bf16 tmp[8];
            if (gk < K && gn + 8 <= N) {
                *reinterpret_cast<uint4*>(tmp) =
                    *reinterpret_cast<const uint4*>(&B[(size_t)gk * ldb + gn]);
                if (k0 + BK < K)
                    __builtin_prefetch(&B[(size_t)(gk + BK) * ldb + gn], 0, 1);
            } else {
                int gkc = (gk < K) ? gk : (K - 1);
                #pragma unroll
                for (int e = 0; e < 8; ++e) {
                    int gne = gn + e;
                    int gnc = (gne < N) ? gne : (N - 1);
                    bf16 v = B[(size_t)gkc * ldb + gnc];
                    tmp[e] = (gk < K && gne < N) ? v : (bf16)0.0f;
                }
            }
            #pragma unroll
            for (int e = 0; e < 8; ++e)
                Bs[(bn_ns + e) * LDB_S + bn_k] = tmp[e];
        } else {
            // logical B(k,n) = Bmat[n*ldb + k]: rows contiguous over k -> b128 read + b128 LDS write
            int gn = colBase + bt_n;
            int gk = k0 + bt_ks;
            bf16 tmp[8];
            if (gn < N && gk + 8 <= K) {
                *reinterpret_cast<uint4*>(tmp) =
                    *reinterpret_cast<const uint4*>(&B[(size_t)gn * ldb + gk]);
            } else {
                int gnc = (gn < N) ? gn : (N - 1);
                #pragma unroll
                for (int e = 0; e < 8; ++e) {
                    int gke = gk + e;
                    int gkc = (gke < K) ? gke : (K - 1);
                    bf16 v = B[(size_t)gnc * ldb + gkc];
                    tmp[e] = (gn < N && gke < K) ? v : (bf16)0.0f;
                }
            }
            *reinterpret_cast<uint4*>(&Bs[bt_n * LDB_S + bt_ks]) = *reinterpret_cast<uint4*>(tmp);
        }
        __syncthreads();

        // ---- fragments + WMMA ----
        v16bf afr[2], bfr[2];
        #pragma unroll
        for (int f = 0; f < 2; ++f) {
            // A layout: lanes 0-15: K=0..7 (v0-3) & 16..23 (v4-7); lanes 16-31: K=8..15 & 24..31
            int m = wm + f * 16 + l16;
            uint4* a128 = reinterpret_cast<uint4*>(&afr[f]);
            a128[0] = *reinterpret_cast<const uint4*>(&As[m * LDA_S + 8 * half]);
            a128[1] = *reinterpret_cast<const uint4*>(&As[m * LDA_S + 16 + 8 * half]);
        }
        #pragma unroll
        for (int f = 0; f < 2; ++f) {
            // B layout: lanes 0-15: K=0..7 (v0-3) & 16..23 (v4-7); lanes 16-31: K=8..15 & 24..31
            int n = wn + f * 16 + l16;
            uint4* b128 = reinterpret_cast<uint4*>(&bfr[f]);
            b128[0] = *reinterpret_cast<const uint4*>(&Bs[n * LDB_S + 8 * half]);
            b128[1] = *reinterpret_cast<const uint4*>(&Bs[n * LDB_S + 16 + 8 * half]);
        }
        #pragma unroll
        for (int i = 0; i < 2; ++i)
            #pragma unroll
            for (int j = 0; j < 2; ++j)
                acc[i][j] = __builtin_amdgcn_wmma_f32_16x16x32_bf16(
                    false, afr[i], false, bfr[j], (short)0, acc[i][j], false, false);
        __syncthreads();
    }

    // ---- epilogue: C layout: VGPR r -> row = r + 8*half, col = lane%16 ----
    #pragma unroll
    for (int i = 0; i < 2; ++i) {
        #pragma unroll
        for (int j = 0; j < 2; ++j) {
            int col = colBase + wn + j * 16 + l16;
            #pragma unroll
            for (int r = 0; r < 8; ++r) {
                int row = rowBase + wm + i * 16 + r + 8 * half;
                if (row < M && col < N) {
                    float v = acc[i][j][r];
                    if (bias) v += bias[col];
                    if (do_gelu) v = gelu_tanh(v);
                    if (res) v += res[(size_t)row * ldres + col];
                    if (outF) outF[(size_t)row * ldc + col] = v;
                    if (outH) outH[(size_t)row * ldc + col] = (bf16)v;
                }
            }
        }
    }
}

// ---------------- elementwise / reduction kernels ----------------
__global__ void f2b(const float* __restrict__ x, bf16* __restrict__ y, int n) {
    for (int i = blockIdx.x * blockDim.x + threadIdx.x; i < n; i += gridDim.x * blockDim.x)
        y[i] = (bf16)x[i];
}

__global__ void patchify(const float* __restrict__ px, bf16* __restrict__ X) {
    int idx = blockIdx.x * blockDim.x + threadIdx.x;  // 1024*1536
    if (idx >= 1024 * 1536) return;
    int row = idx / 1536, col = idx % 1536;
    int hp = row >> 5, wp = row & 31;
    int c = col % 3;
    int t0 = col / 3;
    int pw = t0 % 16; t0 /= 16;
    int ph = t0 % 16; int tp = t0 / 16;
    size_t src = (((size_t)c * 2 + tp) * 512 + hp * 16 + ph) * 512 + wp * 16 + pw;
    X[idx] = (bf16)px[src];
}

__global__ void pos_merge(const float* __restrict__ conv, const float* __restrict__ table,
                          float* __restrict__ hid) {
    int idx = blockIdx.x * blockDim.x + threadIdx.x;  // 1024*1152
    if (idx >= 1024 * 1152) return;
    int rp = idx / 1152, d = idx % 1152;
    int bb = rp & 1, aa = (rp >> 1) & 1, ww = (rp >> 2) & 15, hh = rp >> 6;
    int i = hh * 2 + aa, j = ww * 2 + bb;
    int orig = i * 32 + j;
    float hi = i * (47.0f / 31.0f);
    float wi = j * (47.0f / 31.0f);
    int h0 = (int)floorf(hi); int h1 = (h0 + 1 < 47) ? h0 + 1 : 47;
    int w0 = (int)floorf(wi); int w1 = (w0 + 1 < 47) ? w0 + 1 : 47;
    float dh = hi - h0, dw = wi - w0;
    float pe = (1.0f - dh) * (1.0f - dw) * table[(size_t)(h0 * 48 + w0) * 1152 + d]
             + (1.0f - dh) * dw          * table[(size_t)(h0 * 48 + w1) * 1152 + d]
             + dh * (1.0f - dw)          * table[(size_t)(h1 * 48 + w0) * 1152 + d]
             + dh * dw                   * table[(size_t)(h1 * 48 + w1) * 1152 + d];
    hid[(size_t)rp * 1152 + d] = conv[(size_t)orig * 1152 + d] + pe;
}

__global__ void rope_tab(float* __restrict__ cosb, float* __restrict__ sinb) {
    int idx = blockIdx.x * blockDim.x + threadIdx.x;  // 1024*72
    if (idx >= 1024 * 72) return;
    int rp = idx / 72, d = idx % 72;
    int bb = rp & 1, aa = (rp >> 1) & 1, ww = (rp >> 2) & 15, hh = rp >> 6;
    int hpos = hh * 2 + aa, wpos = ww * 2 + bb;
    int d36 = d % 36;
    int fi; int pos;
    if (d36 < 18) { fi = d36;      pos = hpos; }
    else          { fi = d36 - 18; pos = wpos; }
    float freq = powf(10000.0f, -(float)(2 * fi) / 36.0f);
    float ang = (float)pos * freq;
    cosb[idx] = cosf(ang);
    sinb[idx] = sinf(ang);
}

__global__ void qkv_rope_scatter(const float* __restrict__ qkv,
                                 const float* __restrict__ cosb, const float* __restrict__ sinb,
                                 bf16* __restrict__ qp, bf16* __restrict__ kp, bf16* __restrict__ vp) {
    int idx = blockIdx.x * blockDim.x + threadIdx.x;  // 1024*16*96
    if (idx >= S_TOK * NHEAD * HDIM_P) return;
    int d = idx % HDIM_P; int t = idx / HDIM_P;
    int h = t % NHEAD; int r = t / NHEAD;
    size_t o = ((size_t)h * S_TOK + r) * HDIM_P + d;
    if (d >= HDIM) { qp[o] = (bf16)0.0f; kp[o] = (bf16)0.0f; vp[o] = (bf16)0.0f; return; }
    size_t rb = (size_t)r * (3 * DMODEL);
    float q = qkv[rb + h * HDIM + d];
    float k = qkv[rb + DMODEL + h * HDIM + d];
    float v = qkv[rb + 2 * DMODEL + h * HDIM + d];
    int dro = (d < 36) ? d + 36 : d - 36;
    float sgn = (d < 36) ? -1.0f : 1.0f;
    float qr = sgn * qkv[rb + h * HDIM + dro];
    float kr = sgn * qkv[rb + DMODEL + h * HDIM + dro];
    float c = cosb[r * HDIM + d], s = sinb[r * HDIM + d];
    const float rscale = 0.117851130f;  // 1/sqrt(72), folded into Q
    qp[o] = (bf16)((q * c + qr * s) * rscale);
    kp[o] = (bf16)(k * c + kr * s);
    vp[o] = (bf16)v;
}

__global__ void layernorm_bf16(const float* __restrict__ x, const float* __restrict__ s,
                               const float* __restrict__ b, bf16* __restrict__ y, int cols) {
    int row = blockIdx.x;
    const float* xr = x + (size_t)row * cols;
    __shared__ float red[256];
    float sum = 0.0f, sq = 0.0f;
    for (int c = threadIdx.x; c < cols; c += 256) { float v = xr[c]; sum += v; sq += v * v; }
    red[threadIdx.x] = sum; __syncthreads();
    for (int st = 128; st > 0; st >>= 1) { if (threadIdx.x < st) red[threadIdx.x] += red[threadIdx.x + st]; __syncthreads(); }
    float mu = red[0] / cols; __syncthreads();
    red[threadIdx.x] = sq; __syncthreads();
    for (int st = 128; st > 0; st >>= 1) { if (threadIdx.x < st) red[threadIdx.x] += red[threadIdx.x + st]; __syncthreads(); }
    float var = red[0] / cols - mu * mu;
    float rstd = rsqrtf(var + 1e-6f);
    bf16* yr = y + (size_t)row * cols;
    for (int c = threadIdx.x; c < cols; c += 256)
        yr[c] = (bf16)((xr[c] - mu) * rstd * s[c] + b[c]);
}

__global__ void softmax_rows(const float* __restrict__ x, bf16* __restrict__ p, int cols) {
    int row = blockIdx.x;
    const float* xr = x + (size_t)row * cols;
    __shared__ float red[256];
    float m = -1e30f;
    for (int c = threadIdx.x; c < cols; c += 256) m = fmaxf(m, xr[c]);
    red[threadIdx.x] = m; __syncthreads();
    for (int st = 128; st > 0; st >>= 1) { if (threadIdx.x < st) red[threadIdx.x] = fmaxf(red[threadIdx.x], red[threadIdx.x + st]); __syncthreads(); }
    m = red[0]; __syncthreads();
    float sum = 0.0f;
    for (int c = threadIdx.x; c < cols; c += 256) sum += __expf(xr[c] - m);
    red[threadIdx.x] = sum; __syncthreads();
    for (int st = 128; st > 0; st >>= 1) { if (threadIdx.x < st) red[threadIdx.x] += red[threadIdx.x + st]; __syncthreads(); }
    float inv = 1.0f / red[0];
    bf16* pr = p + (size_t)row * cols;
    for (int c = threadIdx.x; c < cols; c += 256) pr[c] = (bf16)(__expf(xr[c] - m) * inv);
}

// ---------------- host side ----------------
static inline void gemm(hipStream_t st, const bf16* A, int lda, const bf16* B, int ldb, int tB,
                        const float* bias, const float* res, int ldres,
                        float* outF, bf16* outH, int ldc, int M, int N, int K, int gelu) {
    dim3 g((N + BN - 1) / BN, (M + BM - 1) / BM);
    gemm_bf16_wmma<<<g, 256, 0, st>>>(A, lda, B, ldb, tB, bias, res, ldres, outF, outH, ldc, M, N, K, gelu);
}

extern "C" void kernel_launch(void* const* d_in, const int* in_sizes, int n_in,
                              void* d_out, int out_size, void* d_ws, size_t ws_size,
                              hipStream_t stream) {
    (void)in_sizes; (void)n_in; (void)out_size; (void)ws_size;

    const float* pixel       = (const float*)d_in[0];
    const float* conv_kernel = (const float*)d_in[4];
    const float* conv_bias   = (const float*)d_in[5];
    const float* pos_table   = (const float*)d_in[6];
    const float* ln1_s = (const float*)d_in[7];
    const float* ln1_b = (const float*)d_in[8];
    const float* ln2_s = (const float*)d_in[9];
    const float* ln2_b = (const float*)d_in[10];
    const float* qkv_w = (const float*)d_in[11];
    const float* qkv_b = (const float*)d_in[12];
    const float* proj_w = (const float*)d_in[13];
    const float* proj_b = (const float*)d_in[14];
    const float* fc1_w = (const float*)d_in[15];
    const float* fc1_b = (const float*)d_in[16];
    const float* fc2_w = (const float*)d_in[17];
    const float* fc2_b = (const float*)d_in[18];
    const float* mg_ns = (const float*)d_in[19];
    const float* mg_nb = (const float*)d_in[20];
    const float* mg_w1 = (const float*)d_in[21];
    const float* mg_b1 = (const float*)d_in[22];
    const float* mg_w2 = (const float*)d_in[23];
    const float* mg_b2 = (const float*)d_in[24];
    const float* ds_ns = (const float*)d_in[25];
    const float* ds_nb = (const float*)d_in[26];
    const float* ds_w1 = (const float*)d_in[27];
    const float* ds_b1 = (const float*)d_in[28];
    const float* ds_w2 = (const float*)d_in[29];
    const float* ds_b2 = (const float*)d_in[30];
    float* out = (float*)d_out;

    // workspace bump allocator
    char* base = (char*)d_ws;
    size_t off = 0;
    auto alloc = [&](size_t bytes) -> char* {
        char* p = base + off;
        off = (off + bytes + 255) & ~(size_t)255;
        return p;
    };
    float* hid    = (float*)alloc((size_t)S_TOK * DMODEL * 4);          // fp32 residual stream
    bf16*  lnout  = (bf16*)alloc((size_t)S_TOK * DMODEL * 2);           // bf16 LN / activation
    float* gfb    = (float*)alloc((size_t)S_TOK * 3 * DMODEL * 4);      // fp32 GEMM out (qkv / scores / conv)
    float* attn_o = (float*)alloc((size_t)S_TOK * DMODEL * 4);          // fp32 attn concat
    bf16*  hbuf   = (bf16*)alloc((size_t)S_TOK * FFDIM * 2);            // bf16 big activations
    bf16*  Pbuf   = (bf16*)alloc((size_t)S_TOK * S_TOK * 2);            // bf16 softmax probs
    bf16*  qpad   = (bf16*)alloc((size_t)NHEAD * S_TOK * HDIM_P * 2);
    bf16*  kpad   = (bf16*)alloc((size_t)NHEAD * S_TOK * HDIM_P * 2);
    bf16*  vpad   = (bf16*)alloc((size_t)NHEAD * S_TOK * HDIM_P * 2);
    bf16*  wbuf   = (bf16*)alloc((size_t)MERGED * MERGED * 2);          // bf16 weight staging
    float* cosb   = (float*)alloc((size_t)S_TOK * HDIM * 4);
    float* sinb   = (float*)alloc((size_t)S_TOK * HDIM * 4);

    // ---- patch embed: X[1024x1536] @ W[1536x1152] + bias, then +pos & 2x2 merge-perm ----
    patchify<<<(1024 * 1536 + 255) / 256, 256, 0, stream>>>(pixel, hbuf);
    f2b<<<2048, 256, 0, stream>>>(conv_kernel, wbuf, 1536 * DMODEL);
    gemm(stream, hbuf, 1536, wbuf, DMODEL, 0, conv_bias, nullptr, 0,
         gfb, nullptr, DMODEL, S_TOK, DMODEL, 1536, 0);
    pos_merge<<<(S_TOK * DMODEL + 255) / 256, 256, 0, stream>>>(gfb, pos_table, hid);
    rope_tab<<<(S_TOK * HDIM + 255) / 256, 256, 0, stream>>>(cosb, sinb);

    for (int i = 0; i < 8; ++i) {
        // ---- attention ----
        layernorm_bf16<<<S_TOK, 256, 0, stream>>>(hid, ln1_s + i * DMODEL, ln1_b + i * DMODEL, lnout, DMODEL);
        f2b<<<2048, 256, 0, stream>>>(qkv_w + (size_t)i * DMODEL * 3 * DMODEL, wbuf, DMODEL * 3 * DMODEL);
        gemm(stream, lnout, DMODEL, wbuf, 3 * DMODEL, 0, qkv_b + (size_t)i * 3 * DMODEL, nullptr, 0,
             gfb, nullptr, 3 * DMODEL, S_TOK, 3 * DMODEL, DMODEL, 0);
        qkv_rope_scatter<<<(S_TOK * NHEAD * HDIM_P + 255) / 256, 256, 0, stream>>>(
            gfb, cosb, sinb, qpad, kpad, vpad);
        for (int h = 0; h < NHEAD; ++h) {
            const bf16* Qh = qpad + (size_t)h * S_TOK * HDIM_P;
            const bf16* Kh = kpad + (size_t)h * S_TOK * HDIM_P;
            const bf16* Vh = vpad + (size_t)h * S_TOK * HDIM_P;
            // scores = Qh (1024x96) * Kh^T (96x1024)
            gemm(stream, Qh, HDIM_P, Kh, HDIM_P, 1, nullptr, nullptr, 0,
                 gfb, nullptr, S_TOK, S_TOK, S_TOK, HDIM_P, 0);
            softmax_rows<<<S_TOK, 256, 0, stream>>>(gfb, Pbuf, S_TOK);
            // out_h = P (1024x1024) * Vh (1024x72) -> columns h*72..h*72+71 of attn_o
            gemm(stream, Pbuf, S_TOK, Vh, HDIM_P, 0, nullptr, nullptr, 0,
                 attn_o + h * HDIM, nullptr, DMODEL, S_TOK, HDIM, S_TOK, 0);
        }
        f2b<<<2048, 256, 0, stream>>>(attn_o, lnout, S_TOK * DMODEL);
        f2b<<<2048, 256, 0, stream>>>(proj_w + (size_t)i * DMODEL * DMODEL, wbuf, DMODEL * DMODEL);
        gemm(stream, lnout, DMODEL, wbuf, DMODEL, 0, proj_b + (size_t)i * DMODEL, hid, DMODEL,
             hid, nullptr, DMODEL, S_TOK, DMODEL, DMODEL, 0);

        // ---- MLP ----
        layernorm_bf16<<<S_TOK, 256, 0, stream>>>(hid, ln2_s + i * DMODEL, ln2_b + i * DMODEL, lnout, DMODEL);
        f2b<<<2048, 256, 0, stream>>>(fc1_w + (size_t)i * DMODEL * FFDIM, wbuf, DMODEL * FFDIM);
        gemm(stream, lnout, DMODEL, wbuf, FFDIM, 0, fc1_b + (size_t)i * FFDIM, nullptr, 0,
             nullptr, hbuf, FFDIM, S_TOK, FFDIM, DMODEL, 1);
        f2b<<<2048, 256, 0, stream>>>(fc2_w + (size_t)i * FFDIM * DMODEL, wbuf, FFDIM * DMODEL);
        gemm(stream, hbuf, FFDIM, wbuf, DMODEL, 0, fc2_b + (size_t)i * DMODEL, hid, DMODEL,
             hid, nullptr, DMODEL, S_TOK, DMODEL, FFDIM, 0);

        // ---- deepstack merger at layers 2,4,6 ----
        int j = (i == 2) ? 0 : (i == 4) ? 1 : (i == 6) ? 2 : -1;
        if (j >= 0) {
            layernorm_bf16<<<256, 256, 0, stream>>>(hid, ds_ns + (size_t)j * MERGED, ds_nb + (size_t)j * MERGED,
                                                    lnout, MERGED);
            f2b<<<4096, 256, 0, stream>>>(ds_w1 + (size_t)j * MERGED * MERGED, wbuf, MERGED * MERGED);
            gemm(stream, lnout, MERGED, wbuf, MERGED, 0, ds_b1 + (size_t)j * MERGED, nullptr, 0,
                 nullptr, hbuf, MERGED, 256, MERGED, MERGED, 1);
            f2b<<<4096, 256, 0, stream>>>(ds_w2 + (size_t)j * MERGED * OUTD, wbuf, MERGED * OUTD);
            gemm(stream, hbuf, MERGED, wbuf, OUTD, 0, ds_b2 + (size_t)j * OUTD, nullptr, 0,
                 out + (size_t)(1 + j) * 256 * OUTD, nullptr, OUTD, 256, OUTD, MERGED, 0);
        }
    }

    // ---- main merger (LN over D, then view as [256 x 4608]) ----
    layernorm_bf16<<<S_TOK, 256, 0, stream>>>(hid, mg_ns, mg_nb, lnout, DMODEL);
    f2b<<<4096, 256, 0, stream>>>(mg_w1, wbuf, MERGED * MERGED);
    gemm(stream, lnout, MERGED, wbuf, MERGED, 0, mg_b1, nullptr, 0,
         nullptr, hbuf, MERGED, 256, MERGED, MERGED, 1);
    f2b<<<4096, 256, 0, stream>>>(mg_w2, wbuf, MERGED * OUTD);
    gemm(stream, hbuf, MERGED, wbuf, OUTD, 0, mg_b2, nullptr, 0,
         out, nullptr, OUTD, 256, OUTD, MERGED, 0);
}
